// CustomGNN_17093969838500
// MI455X (gfx1250) — compile-verified, matrix operand order checked by reading, not run on hardware
//
#include <hip/hip_runtime.h>

#define Hdim 128
#define TWO_H 256
#define THREE_H 384

typedef __attribute__((ext_vector_type(16))) _Float16 v16h;
typedef __attribute__((ext_vector_type(8)))  float    v8f;

union F8 { v8f v; float f[8]; };

__device__ __forceinline__ v8f wmma_f16(v16h a, v16h b, v8f c) {
  // D = A(16x32 f16) * B(32x16 f16) + C(16x16 f32)
  return __builtin_amdgcn_wmma_f32_16x16x32_f16(false, a, false, b, (short)0, c,
                                                false, false);
}

// Fragment-native packed index for a [K x nCols] matrix, 16-bit elements.
// Layout: for each (kt, col, hi) a contiguous run of 16 halves whose K order
// matches the 16-bit A/B WMMA VGPR layout (ISA 7.12.2):
//   halves 0..7  -> K = kt*32 + hi*8 + j
//   halves 8..15 -> K = kt*32 + 16 + hi*8 + (j-8)
__device__ __forceinline__ int packIdx(int r, int k, int nRows) {
  int kt = k >> 5, kk = k & 31;
  int hi = (kk & 15) >> 3;
  int j  = (kk & 7) + ((kk & 16) ? 8 : 0);
  return ((kt * nRows + r) * 2 + hi) * 16 + j;
}

// ---------------- weight pack: f32 -> f16 fragment layout ----------------
__global__ void pack_weight_kernel(const float* __restrict__ src,
                                   _Float16* __restrict__ dst,
                                   int Ktot, int Ncols, int transposed) {
  int idx = blockIdx.x * blockDim.x + threadIdx.x;
  int total = Ktot * Ncols;
  if (idx >= total) return;
  int j   = idx & 15;
  int hi  = (idx >> 4) & 1;
  int rest = idx >> 5;
  int n   = rest % Ncols;
  int kt  = rest / Ncols;
  int kk  = (j < 8) ? (hi * 8 + j) : (16 + hi * 8 + (j - 8));
  int K   = kt * 32 + kk;
  float v = transposed ? src[n * Ktot + K] : src[K * Ncols + n];
  dst[idx] = (_Float16)v;
}

__global__ void zero_kernel(float* __restrict__ p, int n) {
  int i = blockIdx.x * blockDim.x + threadIdx.x;
  if (i < n) p[i] = 0.f;
}

// ---------------- fused edge MLP + scatter-add ----------------
// 128 edges per block (8 row tiles of 16), 8 waves; wave w owns output cols
// [16w,16w+16). Computes both edge-type MLPs, selects per edge, atomically
// accumulates into messages[tgt].
__global__ __launch_bounds__(256)
void edge_mlp_kernel(const float* __restrict__ h,
                     const int* __restrict__ ei,      // [2,E]
                     const int* __restrict__ etype,   // [E]
                     const _Float16* __restrict__ w1sf, const float* __restrict__ b1sf,
                     const _Float16* __restrict__ w2sf, const float* __restrict__ b2sf,
                     const _Float16* __restrict__ w1fd, const float* __restrict__ b1fd,
                     const _Float16* __restrict__ w2fd, const float* __restrict__ b2fd,
                     float* __restrict__ msg, int E) {
  __shared__ alignas(32) _Float16 efA[8 * 16 * 2 * 16];   // 16 x 256, A-frag layout
  __shared__ alignas(32) _Float16 hsf[4 * 16 * 2 * 16];   // 16 x 128 hidden (sf)
  __shared__ alignas(32) _Float16 hfd[4 * 16 * 2 * 16];   // 16 x 128 hidden (fd)

  const int tid  = threadIdx.x;
  const int wave = tid >> 5;
  const int lane = tid & 31;
  const int col  = lane & 15;   // N (and A-row) index
  const int hi   = lane >> 4;   // K-half select
  const int n0   = wave * 16;

  for (int rt = 0; rt < 8; ++rt) {
    __syncthreads();
    const int eb = (blockIdx.x * 8 + rt) * 16;
    // ---- gather ef = [h[src] | h[tgt]] (f32 -> f16, fragment layout) ----
    {
      int r  = tid >> 4;
      int c0 = (tid & 15) * 16;
      int e  = eb + r;
      int s = 0, t = 0;
      bool valid = (e < E);
      if (valid) { s = ei[e]; t = ei[E + e]; }
      #pragma unroll
      for (int i = 0; i < 16; ++i) {
        int k = c0 + i;
        float v = 0.f;
        if (valid) v = (k < Hdim) ? h[s * Hdim + k] : h[t * Hdim + (k - Hdim)];
        efA[packIdx(r, k, 16)] = (_Float16)v;
      }
    }
    __syncthreads();
    // ---- stage 1: hidden = relu(ef @ W1 + b1), both edge types ----
    {
      F8 asf, afd;
      float bs = b1sf[n0 + col];
      float bf = b1fd[n0 + col];
      #pragma unroll
      for (int g = 0; g < 8; ++g) { asf.f[g] = bs; afd.f[g] = bf; }
      #pragma unroll
      for (int kt = 0; kt < 8; ++kt) {
        v16h a  = *(const v16h*)&efA[((kt * 16 + col) * 2 + hi) * 16];
        v16h b0 = *(const v16h*)&w1sf[((kt * Hdim + n0 + col) * 2 + hi) * 16];
        v16h b1 = *(const v16h*)&w1fd[((kt * Hdim + n0 + col) * 2 + hi) * 16];
        asf.v = wmma_f16(a, b0, asf.v);
        afd.v = wmma_f16(a, b1, afd.v);
      }
      #pragma unroll
      for (int g = 0; g < 8; ++g) {
        int M = g + 8 * hi;
        int kcol = n0 + col;
        float vs = asf.f[g]; vs = vs > 0.f ? vs : 0.f;
        float vf = afd.f[g]; vf = vf > 0.f ? vf : 0.f;
        hsf[packIdx(M, kcol, 16)] = (_Float16)vs;
        hfd[packIdx(M, kcol, 16)] = (_Float16)vf;
      }
    }
    __syncthreads();
    // ---- stage 2: m = hidden @ W2 + b2; select by edge type; scatter ----
    {
      F8 csf, cfd;
      float bs = b2sf[n0 + col];
      float bf = b2fd[n0 + col];
      #pragma unroll
      for (int g = 0; g < 8; ++g) { csf.f[g] = bs; cfd.f[g] = bf; }
      #pragma unroll
      for (int kt = 0; kt < 4; ++kt) {
        v16h a0 = *(const v16h*)&hsf[((kt * 16 + col) * 2 + hi) * 16];
        v16h a1 = *(const v16h*)&hfd[((kt * 16 + col) * 2 + hi) * 16];
        v16h b0 = *(const v16h*)&w2sf[((kt * Hdim + n0 + col) * 2 + hi) * 16];
        v16h b1 = *(const v16h*)&w2fd[((kt * Hdim + n0 + col) * 2 + hi) * 16];
        csf.v = wmma_f16(a0, b0, csf.v);
        cfd.v = wmma_f16(a1, b1, cfd.v);
      }
      #pragma unroll
      for (int g = 0; g < 8; ++g) {
        int M = g + 8 * hi;
        int e = eb + M;
        if (e < E) {
          int tt = ei[E + e];
          float val = (etype[e] == 0) ? csf.f[g] : cfd.f[g];
          atomicAdd(&msg[tt * Hdim + (n0 + col)], val);
        }
      }
    }
  }
}

// ---------------- GRU cell: h' = GRU(messages, h) ----------------
// 16 nodes per block. gi = m @ W_ih^T + b_ih, gh = h @ W_hh^T + b_hh via WMMA
// (weights pre-packed transposed), then fused gate elementwise.
__global__ __launch_bounds__(256)
void gru_kernel(const float* __restrict__ msg, const float* __restrict__ hin,
                const _Float16* __restrict__ wih, const _Float16* __restrict__ whh,
                const float* __restrict__ bih, const float* __restrict__ bhh,
                float* __restrict__ hout, int N) {
  __shared__ alignas(32) _Float16 mA[4 * 16 * 2 * 16];
  __shared__ alignas(32) _Float16 hA[4 * 16 * 2 * 16];
  __shared__ float giL[16 * THREE_H];
  __shared__ float ghL[16 * THREE_H];

  const int tid  = threadIdx.x;
  const int wave = tid >> 5;
  const int lane = tid & 31;
  const int col  = lane & 15;
  const int hi   = lane >> 4;
  const int nb   = blockIdx.x * 16;

  {  // gather + f16 convert into fragment layout
    int r  = tid >> 4;
    int c0 = (tid & 15) * 8;
    int node = nb + r;
    bool valid = node < N;
    #pragma unroll
    for (int i = 0; i < 8; ++i) {
      int k = c0 + i;
      float mv = valid ? msg[node * Hdim + k] : 0.f;
      float hv = valid ? hin[node * Hdim + k] : 0.f;
      int pi = packIdx(r, k, 16);
      mA[pi] = (_Float16)mv;
      hA[pi] = (_Float16)hv;
    }
  }
  __syncthreads();
  #pragma unroll
  for (int s = 0; s < 3; ++s) {
    int n0 = (wave + s * 8) * 16;   // 24 column tiles over 3H=384
    F8 ci, ch;
    float bi = bih[n0 + col];
    float bh = bhh[n0 + col];
    #pragma unroll
    for (int g = 0; g < 8; ++g) { ci.f[g] = bi; ch.f[g] = bh; }
    #pragma unroll
    for (int kt = 0; kt < 4; ++kt) {
      v16h am = *(const v16h*)&mA[((kt * 16 + col) * 2 + hi) * 16];
      v16h ah = *(const v16h*)&hA[((kt * 16 + col) * 2 + hi) * 16];
      v16h bi_v = *(const v16h*)&wih[((kt * THREE_H + n0 + col) * 2 + hi) * 16];
      v16h bh_v = *(const v16h*)&whh[((kt * THREE_H + n0 + col) * 2 + hi) * 16];
      ci.v = wmma_f16(am, bi_v, ci.v);
      ch.v = wmma_f16(ah, bh_v, ch.v);
    }
    #pragma unroll
    for (int g = 0; g < 8; ++g) {
      int M = g + 8 * hi;
      giL[M * THREE_H + n0 + col] = ci.f[g];
      ghL[M * THREE_H + n0 + col] = ch.f[g];
    }
  }
  __syncthreads();
  {  // fused gates
    int r  = tid >> 4;
    int c0 = (tid & 15) * 8;
    int node = nb + r;
    if (node < N) {
      #pragma unroll
      for (int i = 0; i < 8; ++i) {
        int c = c0 + i;
        float gr = giL[r * THREE_H + c]           + ghL[r * THREE_H + c];
        float gz = giL[r * THREE_H + Hdim + c]    + ghL[r * THREE_H + Hdim + c];
        float rr = 1.f / (1.f + __expf(-gr));
        float zz = 1.f / (1.f + __expf(-gz));
        float nn = tanhf(giL[r * THREE_H + 2 * Hdim + c] +
                         rr * ghL[r * THREE_H + 2 * Hdim + c]);
        float ho = hin[node * Hdim + c];
        hout[node * Hdim + c] = (1.f - zz) * nn + zz * ho;
      }
    }
  }
}

// ---------------- readout: softmax(relu(h@Wr1+br1)@Wr2+br2) ----------------
__global__ __launch_bounds__(256)
void readout_kernel(const float* __restrict__ hin,
                    const _Float16* __restrict__ wr1p, const float* __restrict__ br1,
                    const float* __restrict__ wr2, const float* __restrict__ br2,
                    float* __restrict__ out, int N) {
  __shared__ alignas(32) _Float16 hA[4 * 16 * 2 * 16];
  __shared__ float hidL[16 * Hdim];

  const int tid  = threadIdx.x;
  const int wave = tid >> 5;
  const int lane = tid & 31;
  const int col  = lane & 15;
  const int hi   = lane >> 4;
  const int nb   = blockIdx.x * 16;

  {
    int r  = tid >> 4;
    int c0 = (tid & 15) * 8;
    int node = nb + r;
    bool valid = node < N;
    #pragma unroll
    for (int i = 0; i < 8; ++i) {
      int k = c0 + i;
      float hv = valid ? hin[node * Hdim + k] : 0.f;
      hA[packIdx(r, k, 16)] = (_Float16)hv;
    }
  }
  __syncthreads();
  {
    int n0 = wave * 16;
    F8 c;
    float b = br1[n0 + col];
    #pragma unroll
    for (int g = 0; g < 8; ++g) c.f[g] = b;
    #pragma unroll
    for (int kt = 0; kt < 4; ++kt) {
      v16h a = *(const v16h*)&hA[((kt * 16 + col) * 2 + hi) * 16];
      v16h bv = *(const v16h*)&wr1p[((kt * Hdim + n0 + col) * 2 + hi) * 16];
      c.v = wmma_f16(a, bv, c.v);
    }
    #pragma unroll
    for (int g = 0; g < 8; ++g) {
      int M = g + 8 * hi;
      float v = c.f[g];
      hidL[M * Hdim + n0 + col] = v > 0.f ? v : 0.f;
    }
  }
  __syncthreads();
  if (tid < 16) {
    int node = nb + tid;
    if (node < N) {
      float o0 = br2[0], o1 = br2[1];
      for (int k = 0; k < Hdim; ++k) {
        float hv = hidL[tid * Hdim + k];
        o0 += hv * wr2[k * 2 + 0];
        o1 += hv * wr2[k * 2 + 1];
      }
      float m  = fmaxf(o0, o1);
      float e0 = __expf(o0 - m), e1 = __expf(o1 - m);
      float inv = 1.f / (e0 + e1);
      out[node * 2 + 0] = e0 * inv;
      out[node * 2 + 1] = e1 * inv;
    }
  }
}

extern "C" void kernel_launch(void* const* d_in, const int* in_sizes, int n_in,
                              void* d_out, int out_size, void* d_ws, size_t ws_size,
                              hipStream_t stream) {
  const float* x     = (const float*)d_in[0];
  const int*   ei    = (const int*)d_in[1];
  const int*   et    = (const int*)d_in[2];
  const float* w1_sf = (const float*)d_in[3];
  const float* b1_sf = (const float*)d_in[4];
  const float* w2_sf = (const float*)d_in[5];
  const float* b2_sf = (const float*)d_in[6];
  const float* w1_fd = (const float*)d_in[7];
  const float* b1_fd = (const float*)d_in[8];
  const float* w2_fd = (const float*)d_in[9];
  const float* b2_fd = (const float*)d_in[10];
  const float* w_ih  = (const float*)d_in[11];
  const float* w_hh  = (const float*)d_in[12];
  const float* b_ih  = (const float*)d_in[13];
  const float* b_hh  = (const float*)d_in[14];
  const float* wr1   = (const float*)d_in[15];
  const float* br1   = (const float*)d_in[16];
  const float* wr2   = (const float*)d_in[17];
  const float* br2   = (const float*)d_in[18];

  const int N = in_sizes[0] / Hdim;
  const int E = in_sizes[2];

  char* ws = (char*)d_ws;
  size_t off = 0;
  auto wsAlloc = [&](size_t bytes) -> void* {
    void* p = ws + off;
    off += (bytes + 255) & ~(size_t)255;
    return p;
  };
  float* msg = (float*)wsAlloc((size_t)N * Hdim * sizeof(float));
  float* hbA = (float*)wsAlloc((size_t)N * Hdim * sizeof(float));
  float* hbB = (float*)wsAlloc((size_t)N * Hdim * sizeof(float));
  _Float16* w1sf16 = (_Float16*)wsAlloc((size_t)TWO_H * Hdim * 2);
  _Float16* w1fd16 = (_Float16*)wsAlloc((size_t)TWO_H * Hdim * 2);
  _Float16* w2sf16 = (_Float16*)wsAlloc((size_t)Hdim * Hdim * 2);
  _Float16* w2fd16 = (_Float16*)wsAlloc((size_t)Hdim * Hdim * 2);
  _Float16* wih16  = (_Float16*)wsAlloc((size_t)Hdim * THREE_H * 2);
  _Float16* whh16  = (_Float16*)wsAlloc((size_t)Hdim * THREE_H * 2);
  _Float16* wr116  = (_Float16*)wsAlloc((size_t)Hdim * Hdim * 2);
  (void)ws_size; (void)n_in; (void)out_size;

  // pack weights to f16 fragment layout
  {
    int t;
    t = TWO_H * Hdim;
    pack_weight_kernel<<<(t + 255) / 256, 256, 0, stream>>>(w1_sf, w1sf16, TWO_H, Hdim, 0);
    pack_weight_kernel<<<(t + 255) / 256, 256, 0, stream>>>(w1_fd, w1fd16, TWO_H, Hdim, 0);
    t = Hdim * Hdim;
    pack_weight_kernel<<<(t + 255) / 256, 256, 0, stream>>>(w2_sf, w2sf16, Hdim, Hdim, 0);
    pack_weight_kernel<<<(t + 255) / 256, 256, 0, stream>>>(w2_fd, w2fd16, Hdim, Hdim, 0);
    pack_weight_kernel<<<(t + 255) / 256, 256, 0, stream>>>(wr1, wr116, Hdim, Hdim, 0);
    t = Hdim * THREE_H;  // transposed: B[K][n] = W[n][K]
    pack_weight_kernel<<<(t + 255) / 256, 256, 0, stream>>>(w_ih, wih16, Hdim, THREE_H, 1);
    pack_weight_kernel<<<(t + 255) / 256, 256, 0, stream>>>(w_hh, whh16, Hdim, THREE_H, 1);
  }

  const float* hin = x;
  float* houts[3] = { hbA, hbB, hbA };
  const int nm = N * Hdim;
  const int eblocks = (E + 127) / 128;
  const int nblocks = (N + 15) / 16;
  for (int l = 0; l < 3; ++l) {
    zero_kernel<<<(nm + 255) / 256, 256, 0, stream>>>(msg, nm);
    edge_mlp_kernel<<<eblocks, 256, 0, stream>>>(hin, ei, et,
                                                 w1sf16, b1_sf, w2sf16, b2_sf,
                                                 w1fd16, b1_fd, w2fd16, b2_fd,
                                                 msg, E);
    gru_kernel<<<nblocks, 256, 0, stream>>>(msg, hin, wih16, whh16, b_ih, b_hh,
                                            houts[l], N);
    hin = houts[l];
  }
  readout_kernel<<<nblocks, 256, 0, stream>>>(hin, wr116, br1, wr2, br2,
                                              (float*)d_out, N);
}
